// TransformerSpatialAttention_10067403342536
// MI455X (gfx1250) — matrix operand order repaired and checked by reading, not compiled, not run on hardware
//
#include <hip/hip_runtime.h>
#include <hip/hip_bf16.h>

// ---------------------------------------------------------------------------
// Windowed spatial attention (Swin-style) for MI455X / gfx1250, wave32 WMMA.
//
// Math folding: output = sigmoid(mean_c(xa @ W_proj + b_proj)). The channel
// mean collapses the projection to a vector w_bar = W_proj.mean(axis=1), and
// the attention value path collapses to a per-token scalar per head:
//   vb_h[t] = xn[t] . (W_v_h @ w_bar_h) + b_v_h . w_bar_h
// so we only need Q,K GEMMs (f16 WMMA) + scores + streaming softmax-dot.
// ---------------------------------------------------------------------------

typedef __attribute__((ext_vector_type(16))) _Float16 v16h;
typedef __attribute__((ext_vector_type(8)))  float    v8f;

#define CDIM   192
#define NHEAD  8
#define HD     24
#define STOK   64
#define WSZ    8
#define NWIN_X 32      // 256 / 8
#define SCALE  0.2041241452319315f   // 24^-0.5
#define EPS    1e-5f

// workspace layout (floats), then fragment-packed f16 weights
#define WS_WBAR 0          // 192 floats
#define WS_BBAR 192        // 1 float
#define WS_BVB  200        // 8 floats
#define WS_WV   208        // 1536 floats  (wv[h][c])
#define WS_FLT  1792       // float region size (7168 B, 32B aligned)
// wfrag: 384 fragments * 32 lanes * 16 halfs = 196608 halfs (384 KiB)
// fragment id f = ((h*2 + qk)*2 + nt)*6 + ks

// dynamic LDS layout (bytes)
#define LDS_XH    0                    // 64 x 200 f16  = 25600 B
#define XS        200
#define LDS_QK    25600                // 8 waves * 8192 B (q:4096, k:4096)
#define LDS_VB    (25600 + 65536)     // 8*64 f32 = 2048 B
#define LDS_HOUT  (LDS_VB + 2048)     // 8*64 f32 = 2048 B
#define SMEM_BYTES (LDS_HOUT + 2048)  // 95232 B

// ---------------------------------------------------------------------------
// Prep 1: w_bar, b_bar, bvb  (single block)
// ---------------------------------------------------------------------------
__global__ void prep_bar_kernel(const float* __restrict__ w_proj,
                                const float* __restrict__ b_proj,
                                const float* __restrict__ b_qkv,
                                float* __restrict__ wsf) {
  __shared__ float wbar_s[CDIM];
  int tid = threadIdx.x;
  if (tid < CDIM) {
    const float* row = w_proj + (size_t)tid * CDIM;
    float s = 0.f;
    for (int k = 0; k < CDIM; ++k) s += row[k];
    s *= (1.0f / CDIM);
    wbar_s[tid] = s;
    wsf[WS_WBAR + tid] = s;
  }
  __syncthreads();
  if (tid == 0) {
    float s = 0.f;
    for (int k = 0; k < CDIM; ++k) s += b_proj[k];
    wsf[WS_BBAR] = s * (1.0f / CDIM);
  }
  if (tid < NHEAD) {
    float s = 0.f;
    for (int d = 0; d < HD; ++d)
      s += b_qkv[2 * CDIM + tid * HD + d] * wbar_s[tid * HD + d];
    wsf[WS_BVB + tid] = s;
  }
}

// ---------------------------------------------------------------------------
// Prep 2: wv[h][c] = sum_d W_v[c, h*24+d] * w_bar[h*24+d]
// ---------------------------------------------------------------------------
__global__ void prep_wv_kernel(const float* __restrict__ w_qkv,
                               float* __restrict__ wsf) {
  int i = blockIdx.x * blockDim.x + threadIdx.x;
  if (i >= NHEAD * CDIM) return;
  int h = i / CDIM, c = i % CDIM;
  const float* wb = wsf + WS_WBAR + h * HD;
  const float* wr = w_qkv + (size_t)c * (3 * CDIM) + 2 * CDIM + h * HD;
  float s = 0.f;
  for (int d = 0; d < HD; ++d) s += wr[d] * wb[d];
  wsf[WS_WV + i] = s;
}

// ---------------------------------------------------------------------------
// Prep 3: pack Q/K weights into WMMA B-fragment order, f16, N padded 24->32.
// B layout (f16, 32x16): lanes 0-15 hold K=0..7,16..23, lanes 16-31 K=8..15,24..31,
// element j of the 16-half vector = K-row (base+j, or base+8+j for j>=8), column = lane%16.
// ---------------------------------------------------------------------------
__global__ void prep_wfrag_kernel(const float* __restrict__ w_qkv,
                                  _Float16* __restrict__ wfrag) {
  int gid = blockIdx.x * blockDim.x + threadIdx.x;   // 384 frags * 32 lanes
  if (gid >= 384 * 32) return;
  int f = gid >> 5, lane = gid & 31;
  int head = f / 24, rem = f % 24;
  int qk = rem / 12; rem %= 12;
  int nt = rem / 6;
  int ks = rem % 6;
  int kbase = ks * 32 + ((lane & 16) ? 8 : 0);
  int col16 = nt * 16 + (lane & 15);
  _Float16* dst = wfrag + ((size_t)f * 512 + lane * 16);
  for (int j = 0; j < 16; ++j) {
    _Float16 v = (_Float16)0.0f;
    if (col16 < HD) {
      int krow = kbase + (j < 8 ? j : j + 8);
      v = (_Float16)w_qkv[(size_t)krow * (3 * CDIM) + qk * CDIM + head * HD + col16];
    }
    dst[j] = v;
  }
}

// ---------------------------------------------------------------------------
// Main kernel: one block per window, 8 waves = 8 heads.
// ---------------------------------------------------------------------------
__global__ __launch_bounds__(256)
void win_attn_kernel(const float* __restrict__ x,
                     const float* __restrict__ gamma,
                     const float* __restrict__ beta,
                     const float* __restrict__ b_qkv,
                     const float* __restrict__ wsf,
                     const _Float16* __restrict__ wfrag,
                     float* __restrict__ out) {
  extern __shared__ char smem[];
  _Float16* xh = (_Float16*)(smem + LDS_XH);
  float* vb      = (float*)(smem + LDS_VB);
  float* headout = (float*)(smem + LDS_HOUT);

  const int tid  = threadIdx.x;
  const int wave = tid >> 5;          // == head index
  const int lane = tid & 31;
  const int l15  = lane & 15;
  const int lhalf = (lane & 16) ? 1 : 0;

  const int widx = blockIdx.x;
  const int b  = widx >> 10;
  const int wh = (widx >> 5) & 31;
  const int ww = widx & 31;

  // warm the L2->L1 path for this head's packed weights
  {
    const _Float16* wbase = wfrag + (size_t)(wave * 24) * 512 + lane * 16;
    #pragma unroll
    for (int f = 0; f < 24; ++f) __builtin_prefetch(wbase + f * 512, 0, 0);
  }

  // ---- Phase 1: window gather -> f16 token-major tile in LDS -------------
  const float* xbase = x + (size_t)b * CDIM * 65536 + (wh * 8) * 256 + ww * 8;
  for (int i = tid; i < CDIM * WSZ; i += 256) {      // 1536 (c,row) segments
    int c = i >> 3, r = i & 7;
    const float* p = xbase + (size_t)c * 65536 + r * 256;
    float4 v0 = *(const float4*)p;
    float4 v1 = *(const float4*)(p + 4);
    int t = r * 8;
    xh[(t + 0) * XS + c] = (_Float16)v0.x;
    xh[(t + 1) * XS + c] = (_Float16)v0.y;
    xh[(t + 2) * XS + c] = (_Float16)v0.z;
    xh[(t + 3) * XS + c] = (_Float16)v0.w;
    xh[(t + 4) * XS + c] = (_Float16)v1.x;
    xh[(t + 5) * XS + c] = (_Float16)v1.y;
    xh[(t + 6) * XS + c] = (_Float16)v1.z;
    xh[(t + 7) * XS + c] = (_Float16)v1.w;
  }
  __syncthreads();

  // ---- Phase 2: LayerNorm over C, 4 lanes per token ----------------------
  {
    int tok = tid >> 2, qq = tid & 3;
    float s = 0.f, s2 = 0.f;
    for (int c = qq * 48; c < qq * 48 + 48; ++c) {
      float v = (float)xh[tok * XS + c];
      s += v; s2 += v * v;
    }
    s  += __shfl_xor(s, 1);  s  += __shfl_xor(s, 2);
    s2 += __shfl_xor(s2, 1); s2 += __shfl_xor(s2, 2);
    float mu   = s * (1.0f / CDIM);
    float var  = s2 * (1.0f / CDIM) - mu * mu;
    float rstd = rsqrtf(var + EPS);
    for (int c = qq * 48; c < qq * 48 + 48; ++c) {
      float v = (float)xh[tok * XS + c];
      xh[tok * XS + c] = (_Float16)((v - mu) * rstd * gamma[c] + beta[c]);
    }
  }
  __syncthreads();

  // ---- Phase 3: vb[h][t] = xn[t] . wv_h + bvb[h] -------------------------
  for (int idx = tid; idx < NHEAD * STOK; idx += 256) {
    int h = idx >> 6, t = idx & 63;
    float acc = wsf[WS_BVB + h];
    const float* wvh = wsf + WS_WV + h * CDIM;
    for (int c = 0; c < CDIM; ++c) acc += (float)xh[t * XS + c] * wvh[c];
    vb[idx] = acc;
  }
  __syncthreads();

  // ---- Phase 4: Q/K GEMM via v_wmma_f32_16x16x32_f16 ---------------------
  _Float16* qst = (_Float16*)(smem + LDS_QK + wave * 8192);
  _Float16* kst = qst + 2048;   // 64*32 halfs each
  const v16h* wf = (const v16h*)wfrag;
  const float bq0c = b_qkv[wave * HD + l15];
  const float bq1c = (l15 < 8) ? b_qkv[wave * HD + 16 + l15] : 0.f;
  const float bk0c = b_qkv[CDIM + wave * HD + l15];
  const float bk1c = (l15 < 8) ? b_qkv[CDIM + wave * HD + 16 + l15] : 0.f;

  for (int mt = 0; mt < 4; ++mt) {
    v8f aq0 = {}, aq1 = {}, ak0 = {}, ak1 = {};
    int m = mt * 16 + l15;
    for (int ks = 0; ks < 6; ++ks) {
      // A fragment (16x32 f16): per-lane K runs of 8 at kbase and kbase+16
      int kbase = ks * 32 + lhalf * 8;
      const _Float16* ap = &xh[m * XS + kbase];
      v16h a;
      #pragma unroll
      for (int j = 0; j < 8; ++j) { a[j] = ap[j]; a[j + 8] = ap[16 + j]; }
      // B fragments: contiguous 32B per lane from packed workspace
      size_t fb = (size_t)(wave * 24 + ks) * 32 + lane;
      v16h bq0 = wf[fb];             // qk=0, nt=0
      v16h bq1 = wf[fb + 6 * 32];    // qk=0, nt=1
      v16h bk0 = wf[fb + 12 * 32];   // qk=1, nt=0
      v16h bk1 = wf[fb + 18 * 32];   // qk=1, nt=1
      aq0 = __builtin_amdgcn_wmma_f32_16x16x32_f16(false, a, false, bq0, (short)0, aq0, false, false);
      aq1 = __builtin_amdgcn_wmma_f32_16x16x32_f16(false, a, false, bq1, (short)0, aq1, false, false);
      ak0 = __builtin_amdgcn_wmma_f32_16x16x32_f16(false, a, false, bk0, (short)0, ak0, false, false);
      ak1 = __builtin_amdgcn_wmma_f32_16x16x32_f16(false, a, false, bk1, (short)0, ak1, false, false);
    }
    // stage Qh/Kh (d padded to 32, pad zeroed) with bias add
    #pragma unroll
    for (int i = 0; i < 8; ++i) {
      int row = mt * 16 + i + lhalf * 8;
      qst[row * 32 + l15]      = (_Float16)(aq0[i] + bq0c);
      qst[row * 32 + 16 + l15] = (_Float16)((l15 < 8) ? (aq1[i] + bq1c) : 0.f);
      kst[row * 32 + l15]      = (_Float16)(ak0[i] + bk0c);
      kst[row * 32 + 16 + l15] = (_Float16)((l15 < 8) ? (ak1[i] + bk1c) : 0.f);
    }
  }
  __syncthreads();

  // ---- Phase 5: scores (1 WMMA / 16x16 tile, K=32) + fused softmax-dot ---
  const float vbl0 = vb[wave * 64 +  0 + l15];
  const float vbl1 = vb[wave * 64 + 16 + l15];
  const float vbl2 = vb[wave * 64 + 32 + l15];
  const float vbl3 = vb[wave * 64 + 48 + l15];

  for (int mt = 0; mt < 4; ++mt) {
    const _Float16* ap = &qst[(mt * 16 + l15) * 32 + lhalf * 16];
    v16h a;
    #pragma unroll
    for (int j = 0; j < 8; ++j) { a[j] = ap[j]; a[j + 8] = ap[8 + j]; }
    v8f sm[4];
    #pragma unroll
    for (int nt = 0; nt < 4; ++nt) {
      // B = Kh^T fragment: contiguous 16 halfs per lane (row t, d-run)
      v16h bfrag = *(const v16h*)&kst[(nt * 16 + l15) * 32 + lhalf * 16];
      v8f z = {};
      sm[nt] = __builtin_amdgcn_wmma_f32_16x16x32_f16(false, a, false, bfrag, (short)0, z, false, false);
    }
    #pragma unroll
    for (int i = 0; i < 8; ++i) {
      float x0 = sm[0][i] * SCALE, x1 = sm[1][i] * SCALE;
      float x2 = sm[2][i] * SCALE, x3 = sm[3][i] * SCALE;
      float mx = fmaxf(fmaxf(x0, x1), fmaxf(x2, x3));
      mx = fmaxf(mx, __shfl_xor(mx, 1));
      mx = fmaxf(mx, __shfl_xor(mx, 2));
      mx = fmaxf(mx, __shfl_xor(mx, 4));
      mx = fmaxf(mx, __shfl_xor(mx, 8));
      float e0 = __expf(x0 - mx), e1 = __expf(x1 - mx);
      float e2 = __expf(x2 - mx), e3 = __expf(x3 - mx);
      float zz = e0 + e1 + e2 + e3;
      float dd = e0 * vbl0 + e1 * vbl1 + e2 * vbl2 + e3 * vbl3;
      zz += __shfl_xor(zz, 1); dd += __shfl_xor(dd, 1);
      zz += __shfl_xor(zz, 2); dd += __shfl_xor(dd, 2);
      zz += __shfl_xor(zz, 4); dd += __shfl_xor(dd, 4);
      zz += __shfl_xor(zz, 8); dd += __shfl_xor(dd, 8);
      if (l15 == 0)
        headout[wave * 64 + mt * 16 + i + lhalf * 8] = dd / zz;
    }
  }
  __syncthreads();

  // ---- Phase 6: sum heads + b_bar, sigmoid, scatter back ------------------
  if (tid < STOK) {
    float acc = wsf[WS_BBAR];
    #pragma unroll
    for (int h = 0; h < NHEAD; ++h) acc += headout[h * 64 + tid];
    float sg = 1.0f / (1.0f + __expf(-acc));
    int r = tid >> 3, cidx = tid & 7;
    out[(size_t)b * 65536 + (size_t)(wh * 8 + r) * 256 + (ww * 8 + cidx)] = sg;
  }
}

// ---------------------------------------------------------------------------
extern "C" void kernel_launch(void* const* d_in, const int* in_sizes, int n_in,
                              void* d_out, int out_size, void* d_ws, size_t ws_size,
                              hipStream_t stream) {
  const float* x      = (const float*)d_in[0];
  const float* gamma  = (const float*)d_in[1];
  const float* beta   = (const float*)d_in[2];
  const float* w_qkv  = (const float*)d_in[3];
  const float* b_qkv  = (const float*)d_in[4];
  const float* w_proj = (const float*)d_in[5];
  const float* b_proj = (const float*)d_in[6];
  float* out = (float*)d_out;

  float*    wsf   = (float*)d_ws;                       // 7168 B float region
  _Float16* wfrag = (_Float16*)((char*)d_ws + WS_FLT * sizeof(float));
  // workspace use: 7168 + 384*512*2 = 400384 bytes

  (void)in_sizes; (void)n_in; (void)out_size; (void)ws_size;

  hipFuncSetAttribute((const void*)win_attn_kernel,
                      hipFuncAttributeMaxDynamicSharedMemorySize, SMEM_BYTES);

  prep_bar_kernel<<<1, 256, 0, stream>>>(w_proj, b_proj, b_qkv, wsf);
  prep_wv_kernel<<<6, 256, 0, stream>>>(w_qkv, wsf);
  prep_wfrag_kernel<<<48, 256, 0, stream>>>(w_qkv, wfrag);
  win_attn_kernel<<<8192, 256, SMEM_BYTES, stream>>>(
      x, gamma, beta, b_qkv, wsf, wfrag, out);
}
